// GNN_net_59665685676719
// MI455X (gfx1250) — matrix operand order repaired and checked by reading, not compiled, not run on hardware
//
#include <hip/hip_runtime.h>

#define N_NODES  131072
#define N_EDGES  524288
#define N_GRAPHS 4096
#define ND       52
#define ED       14
#define HDIM     256
#define NL       5
#define JKD      1332
#define JKD_P    1344     // JKD padded to K-tile multiple (32)
#define BN_EPS   1e-5f

typedef __attribute__((ext_vector_type(16))) __bf16          v16bf;
typedef __attribute__((ext_vector_type(8)))  float           v8f;
typedef __attribute__((ext_vector_type(16))) unsigned short  v16us;
typedef __attribute__((ext_vector_type(8)))  unsigned short  us8;
typedef __attribute__((ext_vector_type(4)))  float           f4;

__device__ __forceinline__ unsigned short f2bf(float f) {
  union { float f; unsigned int u; } v; v.f = f;
  unsigned int u = v.u;
  u += 0x7FFFu + ((u >> 16) & 1u);        // round-to-nearest-even
  return (unsigned short)(u >> 16);
}

// ---- small utility kernels --------------------------------------------------

__global__ void k_zero(float* __restrict__ p, int n) {
  int i = blockIdx.x * blockDim.x + threadIdx.x;
  if (i < n) p[i] = 0.f;
}

__global__ void k_copy(float* __restrict__ d, const float* __restrict__ s, int n) {
  int i = blockIdx.x * blockDim.x + threadIdx.x;
  if (i < n) d[i] = s[i];
}

// d[rows][dw] = s[rows][sw] zero-padded in cols [sw,dw)
__global__ void k_copy_pad(float* __restrict__ d, const float* __restrict__ s,
                           int rows, int sw, int dw) {
  int i = blockIdx.x * blockDim.x + threadIdx.x;
  if (i < rows * dw) {
    int r = i / dw, c = i - r * dw;
    d[i] = (c < sw) ? s[r * sw + c] : 0.f;
  }
}

// Transposed bf16 weight conversion: src f32 [K][Nc] -> dst bf16 [Nc][Kp], K-pad zeroed.
__global__ void k_cvt_t(const float* __restrict__ src, unsigned short* __restrict__ dst,
                        int K, int Nc, int Kp) {
  int i = blockIdx.x * blockDim.x + threadIdx.x;
  if (i < Nc * Kp) {
    int n = i / Kp, k = i - n * Kp;
    dst[i] = (k < K) ? f2bf(src[k * Nc + n]) : (unsigned short)0;
  }
}

__global__ void k_pool_x(const float* __restrict__ x, const int* __restrict__ batch,
                         float* __restrict__ pooled) {
  int i = blockIdx.x * blockDim.x + threadIdx.x;
  if (i < N_NODES * ND) {
    int n = i / ND, c = i - n * ND;
    atomicAdd(&pooled[(size_t)batch[n] * JKD_P + c], x[i]);
  }
}

// ---- GINE edge path ---------------------------------------------------------
// m = relu(h[src] + edge_attr @ eW + eb); agg[dst] += m. One wave per edge.
__global__ void k_edge_msg(const float* __restrict__ h, int h_ld,
                           const float* __restrict__ edge_attr,
                           const int* __restrict__ edge_index,
                           const float* __restrict__ eW, const float* __restrict__ eb,
                           float* __restrict__ agg, int agg_ld, int fin) {
  __shared__ float sW[ED * HDIM];
  __shared__ float sb[HDIM];
  for (int i = threadIdx.x; i < ED * fin; i += blockDim.x) sW[i] = eW[i];
  for (int i = threadIdx.x; i < fin;      i += blockDim.x) sb[i] = eb[i];
  __syncthreads();
  const int lane   = threadIdx.x & 31;
  const int wave   = blockIdx.x * (blockDim.x >> 5) + (threadIdx.x >> 5);
  const int nwaves = gridDim.x * (blockDim.x >> 5);
  for (int e = wave; e < N_EDGES; e += nwaves) {
    int src = edge_index[e];
    int dst = edge_index[N_EDGES + e];
    float ea = (lane < ED) ? edge_attr[e * ED + lane] : 0.f;
    for (int c = lane; c < fin; c += 32) {
      float msg = sb[c];
#pragma unroll
      for (int k = 0; k < ED; ++k)
        msg = fmaf(__shfl(ea, k, 32), sW[k * fin + c], msg);
      float m = fmaxf(h[(size_t)src * h_ld + c] + msg, 0.f);
      atomicAdd(&agg[(size_t)dst * agg_ld + c], m);
    }
  }
}

// ---- fast WMMA GEMM ---------------------------------------------------------
// out[M][Nc] = epi(A[M][lda(f32)] * Wt^T + bias), Wt: bf16 [Nc][Kp] (zero-padded K).
// Requirements: M % 64 == 0 (grid.y = M/64), Nc % 64 == 0 (grid.x = Nc/64),
// Kp % 32 == 0. 256 threads = 8 waves; wave = 16M x 32N (2 WMMAs / k-step).
template <bool RELU, bool BN, bool POOL>
__global__ void k_gemm_fast(const float* __restrict__ A, int lda, int Kp,
                            const unsigned short* __restrict__ Wt,
                            const float* __restrict__ bias,
                            float* __restrict__ out, int out_ld,
                            const float* __restrict__ bn_g, const float* __restrict__ bn_b,
                            const float* __restrict__ bn_m, const float* __restrict__ bn_v,
                            float* __restrict__ pooled, const int* __restrict__ batch,
                            int pool_off) {
  __shared__ __align__(16) unsigned short As[64][40];   // [m][k] bf16 (80B row, 16B aligned)
  __shared__ __align__(16) unsigned short Bst[64][40];  // [n][k] bf16
  const int tid   = threadIdx.x;
  const int lane  = tid & 31;
  const int wv    = tid >> 5;         // 0..7
  const int mt    = wv & 3;           // 16-row tile
  const int nt    = wv >> 2;          // 32-col half
  const int bm    = blockIdx.y * 64;
  const int bnn   = blockIdx.x * 64;
  const int sr    = tid >> 2;         // stage row 0..63
  const int sq    = tid & 3;          // stage 8-elem chunk 0..3

  v8f acc0 = {}, acc1 = {};
  for (int k0 = 0; k0 < Kp; k0 += 32) {
    // stage A: 64x32 f32 -> bf16, two b128 loads + one b128 LDS store per thread
    {
      const float* ap = A + (size_t)(bm + sr) * lda + k0 + sq * 8;
      f4 x0 = *(const f4*)ap;
      f4 x1 = *(const f4*)(ap + 4);
      us8 v;
      v[0] = f2bf(x0[0]); v[1] = f2bf(x0[1]); v[2] = f2bf(x0[2]); v[3] = f2bf(x0[3]);
      v[4] = f2bf(x1[0]); v[5] = f2bf(x1[1]); v[6] = f2bf(x1[2]); v[7] = f2bf(x1[3]);
      *(us8*)&As[sr][sq * 8] = v;
    }
    // stage B (already bf16, transposed): one b128 load + one b128 LDS store
    *(us8*)&Bst[sr][sq * 8] =
        *(const us8*)(Wt + (size_t)(bnn + sr) * Kp + k0 + sq * 8);
    __syncthreads();

    // A fragment (ISA 7.12.2 16-bit 16x32): lane<16 K {0-7,16-23}, lane>=16 K {8-15,24-31}
    const int mrow  = mt * 16 + (lane & 15);
    const int khalf = lane >> 4;
    us8 alo = *(const us8*)&As[mrow][khalf * 8];
    us8 ahi = *(const us8*)&As[mrow][16 + khalf * 8];
    v16us au = __builtin_shufflevector(alo, ahi, 0, 1, 2, 3, 4, 5, 6, 7,
                                       8, 9, 10, 11, 12, 13, 14, 15);
    v16bf a = __builtin_bit_cast(v16bf, au);

    // B fragments: column n holds K-contiguous data in Bst[n][*]
    const int nc0 = nt * 32 + (lane & 15);
    const int kb  = khalf * 8;
    us8 b0lo = *(const us8*)&Bst[nc0][kb];
    us8 b0hi = *(const us8*)&Bst[nc0][kb + 16];
    us8 b1lo = *(const us8*)&Bst[nc0 + 16][kb];
    us8 b1hi = *(const us8*)&Bst[nc0 + 16][kb + 16];
    v16us bu0 = __builtin_shufflevector(b0lo, b0hi, 0, 1, 2, 3, 4, 5, 6, 7,
                                        8, 9, 10, 11, 12, 13, 14, 15);
    v16us bu1 = __builtin_shufflevector(b1lo, b1hi, 0, 1, 2, 3, 4, 5, 6, 7,
                                        8, 9, 10, 11, 12, 13, 14, 15);
    acc0 = __builtin_amdgcn_wmma_f32_16x16x32_bf16(
        false, a, false, __builtin_bit_cast(v16bf, bu0), (short)0, acc0, false, false);
    acc1 = __builtin_amdgcn_wmma_f32_16x16x32_bf16(
        false, a, false, __builtin_bit_cast(v16bf, bu1), (short)0, acc1, false, false);
    __syncthreads();
  }

  // C/D layout: lane%16 = col, lanes>=16 rows +8; vgpr r = row within 8
  const int row0 = bm + mt * 16 + (lane >> 4) * 8;
#pragma unroll
  for (int s = 0; s < 2; ++s) {
    const int col = bnn + nt * 32 + s * 16 + (lane & 15);
    v8f accv = s ? acc1 : acc0;
    float bv = bias[col];
    float scale = 1.f, mean = 0.f, beta = 0.f;
    if (BN) {
      scale = bn_g[col] * rsqrtf(bn_v[col] + BN_EPS);
      mean  = bn_m[col];
      beta  = bn_b[col];
    }
#pragma unroll
    for (int r = 0; r < 8; ++r) {
      const int row = row0 + r;
      float val = accv[r] + bv;
      if (BN) val = (val - mean) * scale + beta;
      if (RELU) val = fmaxf(val, 0.f);
      out[(size_t)row * out_ld + col] = val;
      if (POOL)
        atomicAdd(&pooled[(size_t)batch[row] * JKD_P + pool_off + col], val);
    }
  }
}

// ---- tiny prediction heads (25 MFLOP total -> scalar, weights in LDS) -------
__global__ void k_heads(const float* __restrict__ g2,
                        const float* __restrict__ w1a, const float* __restrict__ b1a,
                        const float* __restrict__ w2a, const float* __restrict__ b2a,
                        const float* __restrict__ w1b, const float* __restrict__ b1b,
                        const float* __restrict__ w2b, const float* __restrict__ b2b,
                        const float* __restrict__ w1c, const float* __restrict__ b1c,
                        const float* __restrict__ w2c, const float* __restrict__ b2c,
                        float* __restrict__ out) {
  __shared__ float sW1[3][64 * 32];
  __shared__ float sb1[3][32];
  __shared__ float sW2[3][32];
  __shared__ float sb2[3];
  const float* w1s[3] = {w1a, w1b, w1c};
  const float* b1s[3] = {b1a, b1b, b1c};
  const float* w2s[3] = {w2a, w2b, w2c};
  const float* b2s[3] = {b2a, b2b, b2c};
  for (int t = 0; t < 3; ++t) {
    for (int i = threadIdx.x; i < 64 * 32; i += blockDim.x) sW1[t][i] = w1s[t][i];
    if (threadIdx.x < 32) sb1[t][threadIdx.x] = b1s[t][threadIdx.x];
    if (threadIdx.x < 32) sW2[t][threadIdx.x] = w2s[t][threadIdx.x];
    if (threadIdx.x == 0) sb2[t] = b2s[t][0];
  }
  __syncthreads();
  const int g = blockIdx.x * blockDim.x + threadIdx.x;
  if (g < N_GRAPHS) {
    float r[64];
#pragma unroll
    for (int k = 0; k < 64; ++k) r[k] = g2[(size_t)g * 64 + k];
    for (int t = 0; t < 3; ++t) {
      float acc = sb2[t];
      for (int j = 0; j < 32; ++j) {
        float s = sb1[t][j];
#pragma unroll
        for (int k = 0; k < 64; ++k) s = fmaf(r[k], sW1[t][k * 32 + j], s);
        acc = fmaf(fmaxf(s, 0.f), sW2[t][j], acc);
      }
      out[(size_t)g * 3 + t] = acc;
    }
  }
}

// ---- host launch ------------------------------------------------------------

static inline dim3 fgrid(int M, int Nc) { return dim3(Nc / 64, M / 64); }

extern "C" void kernel_launch(void* const* d_in, const int* in_sizes, int n_in,
                              void* d_out, int out_size, void* d_ws, size_t ws_size,
                              hipStream_t stream) {
  (void)in_sizes; (void)n_in; (void)out_size; (void)ws_size;
  // input map (setup_inputs dict order, leaves flattened)
  const float* x          = (const float*)d_in[0];
  const int*   edge_index = (const int*)  d_in[1];
  const float* edge_attr  = (const float*)d_in[2];
  const int*   batch      = (const int*)  d_in[3];
  const float *eW[NL], *eb[NL];
  for (int l = 0; l < NL; ++l) { eW[l] = (const float*)d_in[4 + 2 * l]; eb[l] = (const float*)d_in[5 + 2 * l]; }
  const float *W1[NL], *b1[NL], *W2[NL], *b2[NL];
  for (int l = 0; l < NL; ++l) {
    W1[l] = (const float*)d_in[14 + 4 * l]; b1[l] = (const float*)d_in[15 + 4 * l];
    W2[l] = (const float*)d_in[16 + 4 * l]; b2[l] = (const float*)d_in[17 + 4 * l];
  }
  const float *bg[NL], *bbt[NL], *bmn[NL], *bvr[NL];
  for (int l = 0; l < NL; ++l) {
    bg[l]  = (const float*)d_in[34 + 4 * l]; bbt[l] = (const float*)d_in[35 + 4 * l];
    bmn[l] = (const float*)d_in[36 + 4 * l]; bvr[l] = (const float*)d_in[37 + 4 * l];
  }
  const float* fW1 = (const float*)d_in[54]; const float* fb1 = (const float*)d_in[55];
  const float* fW2 = (const float*)d_in[56]; const float* fb2 = (const float*)d_in[57];
  const float *hW1[3], *hb1[3], *hW2[3], *hb2[3];
  for (int t = 0; t < 3; ++t) {
    hW1[t] = (const float*)d_in[58 + 4 * t]; hb1[t] = (const float*)d_in[59 + 4 * t];
    hW2[t] = (const float*)d_in[60 + 4 * t]; hb2[t] = (const float*)d_in[61 + 4 * t];
  }
  float* outp = (float*)d_out;

  // workspace carve
  char* wsp = (char*)d_ws;
  auto take = [&](size_t bytes) -> char* {
    char* p = wsp; wsp += (bytes + 255) & ~(size_t)255; return p;
  };
  float* P      = (float*)take((size_t)N_NODES * HDIM * 4);   // ping
  float* Q      = (float*)take((size_t)N_NODES * HDIM * 4);   // pong
  float* pooled = (float*)take((size_t)N_GRAPHS * JKD_P * 4);
  float* g1     = (float*)take((size_t)N_GRAPHS * 128 * 4);
  float* g2     = (float*)take((size_t)N_GRAPHS * 64 * 4);

  // transposed, K-padded bf16 weights
  auto cvt_t = [&](const float* src, int K, int Nc, int Kp) -> unsigned short* {
    unsigned short* d = (unsigned short*)take((size_t)Nc * Kp * 2);
    int n = Nc * Kp;
    k_cvt_t<<<(n + 255) / 256, 256, 0, stream>>>(src, d, K, Nc, Kp);
    return d;
  };
  unsigned short *bW1t[NL], *bW2t[NL];
  for (int l = 0; l < NL; ++l) {
    int fin  = (l == 0) ? ND : HDIM;
    int finp = (l == 0) ? 64 : HDIM;
    bW1t[l] = cvt_t(W1[l], fin, HDIM, finp);
    bW2t[l] = cvt_t(W2[l], HDIM, HDIM, HDIM);
  }
  unsigned short* bfW1t = cvt_t(fW1, JKD, 128, JKD_P);
  unsigned short* bfW2t = cvt_t(fW2, 128, 64, 128);

  // pooled = 0 (incl. K-pad cols, read as zeros by FFN GEMM); pool x into cols [0,52)
  { int n = N_GRAPHS * JKD_P; k_zero<<<(n + 255) / 256, 256, 0, stream>>>(pooled, n); }
  { int n = N_NODES * ND; k_pool_x<<<(n + 255) / 256, 256, 0, stream>>>(x, batch, pooled); }

  // ---- GINE layers ----
  for (int l = 0; l < NL; ++l) {
    const int fin  = (l == 0) ? ND : HDIM;
    const int finp = (l == 0) ? 64 : HDIM;     // padded K for GEMM1
    float* A0 = (l % 2 == 0) ? P : Q;          // agg, later h_cur
    float* T  = (l % 2 == 0) ? Q : P;          // MLP hidden
    const float* hp  = (l == 0) ? x : T;       // previous h lives in T's buffer
    const int    hld = (l == 0) ? ND : HDIM;
    // agg = h (z = h + sum of messages), stored K-padded
    if (l == 0) {
      int n = N_NODES * 64;
      k_copy_pad<<<(n + 255) / 256, 256, 0, stream>>>(A0, x, N_NODES, ND, 64);
    } else {
      int n = N_NODES * HDIM;
      k_copy<<<(n + 255) / 256, 256, 0, stream>>>(A0, hp, n);
    }
    k_edge_msg<<<2048, 256, 0, stream>>>(hp, hld, edge_attr, edge_index,
                                         eW[l], eb[l], A0, finp, fin);
    // T = relu(agg @ W1 + b1)
    k_gemm_fast<true, false, false><<<fgrid(N_NODES, HDIM), 256, 0, stream>>>(
        A0, finp, finp, bW1t[l], b1[l], T, HDIM,
        nullptr, nullptr, nullptr, nullptr, nullptr, nullptr, 0);
    // h_cur = BN(T @ W2 + b2) [relu if l<4], fused JK sum-pool
    if (l < NL - 1)
      k_gemm_fast<true, true, true><<<fgrid(N_NODES, HDIM), 256, 0, stream>>>(
          T, HDIM, HDIM, bW2t[l], b2[l], A0, HDIM,
          bg[l], bbt[l], bmn[l], bvr[l], pooled, batch, ND + l * HDIM);
    else
      k_gemm_fast<false, true, true><<<fgrid(N_NODES, HDIM), 256, 0, stream>>>(
          T, HDIM, HDIM, bW2t[l], b2[l], A0, HDIM,
          bg[l], bbt[l], bmn[l], bvr[l], pooled, batch, ND + l * HDIM);
  }

  // ---- graph-level FFN ----
  k_gemm_fast<true, false, false><<<fgrid(N_GRAPHS, 128), 256, 0, stream>>>(
      pooled, JKD_P, JKD_P, bfW1t, fb1, g1, 128,
      nullptr, nullptr, nullptr, nullptr, nullptr, nullptr, 0);
  k_gemm_fast<true, false, false><<<fgrid(N_GRAPHS, 64), 256, 0, stream>>>(
      g1, 128, 128, bfW2t, fb2, g2, 64,
      nullptr, nullptr, nullptr, nullptr, nullptr, nullptr, 0);

  // ---- heads -> d_out [G,3] ----
  k_heads<<<N_GRAPHS / 256, 256, 0, stream>>>(
      g2, hW1[0], hb1[0], hW2[0], hb2[0],
      hW1[1], hb1[1], hW2[1], hb2[1],
      hW1[2], hb1[2], hW2[2], hb2[2], outp);
}
